// LSTM_Decoder_90933047591311
// MI455X (gfx1250) — matrix operand order repaired and checked by reading, not compiled, not run on hardware
//
#include <hip/hip_runtime.h>

// ---------------------------------------------------------------------------
// LSTM autoregressive decoder for MI455X (gfx1250), bf16 WMMA path with
// async-to-LDS weight staging (CDNA5 ASYNCcnt pipeline).
// B=1024, LATENT=64, F=128, U=1024, T=256.
// ---------------------------------------------------------------------------

#define BATCH   1024
#define LATENT  64
#define NF      128
#define UNITS   1024
#define TSTEPS  256

typedef __attribute__((ext_vector_type(16))) __bf16 v16bf;
typedef __attribute__((ext_vector_type(8)))  float  v8f;

union FragCast {
    v16bf v;
    uint4 q[2];
};

__device__ __forceinline__ unsigned short f2bf(float f) {
    union { float f; unsigned u; } x;
    x.f = f;
    unsigned u = x.u;
    u += 0x7fffu + ((u >> 16) & 1u);   // round-to-nearest-even
    return (unsigned short)(u >> 16);
}

__device__ __forceinline__ float sigmoidf_dev(float x) {
    return 1.0f / (1.0f + __expf(-x));
}

// A fragment (16x32 bf16, row-major source).  Lane holds row = tileRow + (lane&15),
// K chunks [k0..k0+7] and [k0+8..k0+15] where k0 = (lane>>4)*16 (caller folds in).
__device__ __forceinline__ v16bf load_frag_a(const unsigned short* rowp, int k) {
    FragCast f;
    f.q[0] = *(const uint4*)(rowp + k);
    f.q[1] = *(const uint4*)(rowp + k + 8);
    return f.v;
}

// B fragment from a lane-contiguous 512-bf16 tile (global or LDS pointer).
__device__ __forceinline__ v16bf load_frag_b(const unsigned short* tilep, int lane) {
    FragCast f;
    const uint4* p = (const uint4*)(tilep + lane * 16);
    f.q[0] = p[0];
    f.q[1] = p[1];
    return f.v;
}

__device__ __forceinline__ v8f wmma_bf16(v16bf a, v16bf b, v8f c) {
    return __builtin_amdgcn_wmma_f32_16x16x32_bf16(
        /*neg_a=*/false, a, /*neg_b=*/false, b,
        /*c_mod=*/(short)0, c, /*reuse_a=*/false, /*reuse_b=*/false);
}

// Async copy 16 bytes global -> LDS (per lane).  Tracked by ASYNCcnt.
__device__ __forceinline__ void async_copy_b128(unsigned lds_off, const void* gsrc) {
    asm volatile("global_load_async_to_lds_b128 %0, %1, off"
                 :: "v"(lds_off), "v"((unsigned long long)(uintptr_t)gsrc)
                 : "memory");
}

__device__ __forceinline__ void wait_asynccnt0() {
    asm volatile("s_wait_asynccnt 0x0" ::: "memory");
}

// ---------------------------------------------------------------------------
// Pack fp32 weight W[K][N] (row-major) into bf16 WMMA-B fragments.
// Tile (kt, nt): packed[tile*512 + lane*16 + j] = W[kt*32 + (lane>>4)*16 + j][nt*16 + (lane&15)]
// ---------------------------------------------------------------------------
__global__ void pack_b_bf16_kernel(const float* __restrict__ W,
                                   unsigned short* __restrict__ P,
                                   int K, int N) {
    int id = blockIdx.x * blockDim.x + threadIdx.x;
    int total = K * N;
    if (id >= total) return;
    int NT = N >> 4;
    int tile = id >> 9;
    int within = id & 511;
    int lane = within >> 4;
    int j = within & 15;
    int kt = tile / NT;
    int nt = tile - kt * NT;
    int k = kt * 32 + (lane >> 4) * 16 + j;
    int n = nt * 16 + (lane & 15);
    P[id] = f2bf(W[(size_t)k * N + n]);
}

__global__ void zero_u32_kernel(unsigned int* __restrict__ p, int n) {
    int id = blockIdx.x * blockDim.x + threadIdx.x;
    if (id < n) p[id] = 0u;
}

// x0 = x @ W_in + b_in  -> bf16 feedback buffer.  [1024x64]@[64x128], cheap.
__global__ void in_dense_kernel(const float* __restrict__ x,
                                const float* __restrict__ W_in,
                                const float* __restrict__ b_in,
                                unsigned short* __restrict__ xb) {
    int id = blockIdx.x * blockDim.x + threadIdx.x;   // BATCH*NF threads
    int b = id >> 7;
    int f = id & (NF - 1);
    float s = b_in[f];
    const float* xr = x + (size_t)b * LATENT;
    #pragma unroll 8
    for (int k = 0; k < LATENT; ++k)
        s += xr[k] * W_in[(size_t)k * NF + f];
    xb[id] = f2bf(s);
}

// ---------------------------------------------------------------------------
// Gate kernel: block = 256 threads = 8 waves, ALL sharing one U-tile (ut) but
// covering 8 consecutive M-tiles.  Per K-tile the block stages the 4 gate
// B-fragment tiles (4 KB) into LDS with async-to-LDS copies, double-buffered
// so the copy of tile k+1 overlaps the 4 WMMAs of tile k.  B-side L2 traffic
// drops 8x vs per-wave loads.  C/D layout used for the fused gate epilogue.
// Grid: 64 ut * 8 m-groups = 512 blocks.
// ---------------------------------------------------------------------------
__global__ void lstm_gate_kernel(const unsigned short* __restrict__ xb,
                                 const unsigned short* __restrict__ packK,
                                 const unsigned short* __restrict__ packR,
                                 const float* __restrict__ bias,
                                 const unsigned short* __restrict__ h_in,
                                 unsigned short* __restrict__ h_out,
                                 float* __restrict__ c) {
    __shared__ __align__(16) unsigned short Bst[2][4 * 512];   // 8 KB, double-buffered

    const int tid  = threadIdx.x;
    const int lane = tid & 31;
    const int wv   = tid >> 5;               // 0..7
    const int ut   = blockIdx.x & 63;        // shared by all waves in block
    const int mt   = (blockIdx.x >> 6) * 8 + wv;

    // Staging role of this thread: gate g, 16B chunk within the gate tile.
    const int sg = tid >> 6;                 // 0..3
    const int sc = tid & 63;                 // 0..63 (16B chunks)
    const unsigned lds_dst0 = (unsigned)(uintptr_t)&Bst[0][sg * 512 + sc * 8];
    const unsigned lds_dst1 = (unsigned)(uintptr_t)&Bst[1][sg * 512 + sc * 8];

    // 36 K-tiles total: 0..3 from packK (K=128), 4..35 from packR (K=1024).
    auto tile_src = [&](int i) -> const unsigned short* {
        const unsigned short* base = (i < 4)
            ? packK + (size_t)(i * 256 + ut) * 512
            : packR + (size_t)((i - 4) * 256 + ut) * 512;
        return base + (size_t)sg * (64 * 512) + sc * 8;
    };

    v8f acc0 = {}, acc1 = {}, acc2 = {}, acc3 = {};
    const int row = mt * 16 + (lane & 15);
    const int k0  = (lane >> 4) * 16;
    const unsigned short* xr = xb + (size_t)row * NF;
    const unsigned short* hr = h_in + (size_t)row * UNITS;

    // Prologue: stage tile 0 into buffer 0.
    async_copy_b128(lds_dst0, tile_src(0));
    wait_asynccnt0();
    __syncthreads();

    for (int i = 0; i < 36; ++i) {
        const int cur = i & 1;
        // Kick off async staging of the next tile into the other buffer
        // (safe: that buffer's readers finished at the end-of-loop barrier).
        if (i + 1 < 36)
            async_copy_b128(cur ? lds_dst0 : lds_dst1, tile_src(i + 1));

        // A fragment for this K-tile (L2-resident activations).
        v16bf a = (i < 4) ? load_frag_a(xr, i * 32 + k0)
                          : load_frag_a(hr, (i - 4) * 32 + k0);

        const unsigned short* bb = &Bst[cur][0];
        acc0 = wmma_bf16(a, load_frag_b(bb + 0 * 512, lane), acc0);
        acc1 = wmma_bf16(a, load_frag_b(bb + 1 * 512, lane), acc1);
        acc2 = wmma_bf16(a, load_frag_b(bb + 2 * 512, lane), acc2);
        acc3 = wmma_bf16(a, load_frag_b(bb + 3 * 512, lane), acc3);

        // Next-tile staging complete + all waves done reading `cur`.
        wait_asynccnt0();
        __syncthreads();
    }

    // C/D layout: VGPR r, lane l -> M = mt*16 + r + (l>>4)*8, N = ut*16 + (l&15).
    const int u  = ut * 16 + (lane & 15);
    const float bi = bias[u];
    const float bf = bias[UNITS + u];
    const float bg = bias[2 * UNITS + u];
    const float bo = bias[3 * UNITS + u];
    const int mbase = mt * 16 + (lane >> 4) * 8;

    #pragma unroll
    for (int r = 0; r < 8; ++r) {
        int m = mbase + r;
        size_t idx = (size_t)m * UNITS + u;
        float ig = sigmoidf_dev(acc0[r] + bi);
        float fg = sigmoidf_dev(acc1[r] + bf);
        float gg = tanhf(acc2[r] + bg);
        float og = sigmoidf_dev(acc3[r] + bo);
        float cn = fg * c[idx] + ig * gg;
        c[idx] = cn;
        float hn = og * tanhf(cn);
        h_out[idx] = f2bf(hn);
    }
}

// ---------------------------------------------------------------------------
// Output kernel: y = tanh(h @ W_out + b_out); writes out[:, t, :] fp32 and
// the bf16 feedback buffer for the next timestep.
// Grid: 64 M-tiles * 8 F-tiles = 512 waves; block = 128 threads.
// ---------------------------------------------------------------------------
__global__ void lstm_out_kernel(const unsigned short* __restrict__ h,
                                const unsigned short* __restrict__ packW,
                                const float* __restrict__ b_out,
                                float* __restrict__ out, int t,
                                unsigned short* __restrict__ xb) {
    int wave = (blockIdx.x * blockDim.x + threadIdx.x) >> 5;
    int lane = threadIdx.x & 31;
    int mt = wave >> 3;
    int nt = wave & 7;

    v8f acc = {};
    int row = mt * 16 + (lane & 15);
    int k0  = (lane >> 4) * 16;
    const unsigned short* hr = h + (size_t)row * UNITS;

    #pragma unroll 4
    for (int kt = 0; kt < 32; ++kt) {
        v16bf a = load_frag_a(hr, kt * 32 + k0);
        v16bf b = load_frag_b(packW + (size_t)(kt * 8 + nt) * 512, lane);
        acc = wmma_bf16(a, b, acc);
    }

    int f = nt * 16 + (lane & 15);
    float bo = b_out[f];
    int mbase = mt * 16 + (lane >> 4) * 8;

    #pragma unroll
    for (int r = 0; r < 8; ++r) {
        int m = mbase + r;
        float y = tanhf(acc[r] + bo);
        out[(size_t)m * (TSTEPS * NF) + (size_t)t * NF + f] = y;
        xb[(size_t)m * NF + f] = f2bf(y);
    }
}

// ---------------------------------------------------------------------------
extern "C" void kernel_launch(void* const* d_in, const int* in_sizes, int n_in,
                              void* d_out, int out_size, void* d_ws, size_t ws_size,
                              hipStream_t stream) {
    const float* x          = (const float*)d_in[0];
    const float* W_in       = (const float*)d_in[1];
    const float* b_in       = (const float*)d_in[2];
    const float* kernel     = (const float*)d_in[3];
    const float* rec_kernel = (const float*)d_in[4];
    const float* bias       = (const float*)d_in[5];
    const float* W_out      = (const float*)d_in[6];
    const float* b_out      = (const float*)d_in[7];
    float* out = (float*)d_out;

    // Workspace carve-up (256B aligned).
    char* ws = (char*)d_ws;
    size_t off = 0;
    auto carve = [&](size_t bytes) {
        char* p = ws + off;
        off = (off + bytes + 255) & ~(size_t)255;
        return p;
    };
    unsigned short* packK = (unsigned short*)carve((size_t)NF * 4 * UNITS * 2);      // 1 MB
    unsigned short* packR = (unsigned short*)carve((size_t)UNITS * 4 * UNITS * 2);   // 8 MB
    unsigned short* packW = (unsigned short*)carve((size_t)UNITS * NF * 2);          // 256 KB
    unsigned short* h0    = (unsigned short*)carve((size_t)BATCH * UNITS * 2);       // 2 MB
    unsigned short* h1    = (unsigned short*)carve((size_t)BATCH * UNITS * 2);       // 2 MB
    float*          cbuf  = (float*)carve((size_t)BATCH * UNITS * 4);                // 4 MB
    unsigned short* xbuf  = (unsigned short*)carve((size_t)BATCH * NF * 2);          // 256 KB

    // One-time (per call) weight packing fp32 -> bf16 WMMA fragments.
    {
        int n = NF * 4 * UNITS;                     // 524288
        pack_b_bf16_kernel<<<(n + 255) / 256, 256, 0, stream>>>(kernel, packK, NF, 4 * UNITS);
    }
    {
        int n = UNITS * 4 * UNITS;                  // 4194304
        pack_b_bf16_kernel<<<(n + 255) / 256, 256, 0, stream>>>(rec_kernel, packR, UNITS, 4 * UNITS);
    }
    {
        int n = UNITS * NF;                         // 131072
        pack_b_bf16_kernel<<<(n + 255) / 256, 256, 0, stream>>>(W_out, packW, UNITS, NF);
    }

    // Zero initial h and c (deterministic per call; harness never re-poisons).
    {
        int n = BATCH * UNITS / 2;                  // h0: 2MB -> 524288 u32
        zero_u32_kernel<<<(n + 255) / 256, 256, 0, stream>>>((unsigned int*)h0, n);
        int m = BATCH * UNITS;                      // c: 4MB -> 1048576 u32
        zero_u32_kernel<<<(m + 255) / 256, 256, 0, stream>>>((unsigned int*)cbuf, m);
    }

    // x0 = x @ W_in + b_in.
    {
        int n = BATCH * NF;                         // 131072
        in_dense_kernel<<<(n + 255) / 256, 256, 0, stream>>>(x, W_in, b_in, xbuf);
    }

    // 256 sequential timesteps; ping-pong h buffers to avoid RAW races across
    // tiles inside the gate GEMM.
    dim3 gateGrid(512), gateBlk(256);               // 8 waves/block share one ut
    dim3 outGrid(128),  outBlk(128);                // 512 waves
    for (int t = 0; t < TSTEPS; ++t) {
        const unsigned short* hin  = (t & 1) ? h1 : h0;
        unsigned short*       hout = (t & 1) ? h0 : h1;
        lstm_gate_kernel<<<gateGrid, gateBlk, 0, stream>>>(
            xbuf, packK, packR, bias, hin, hout, cbuf);
        lstm_out_kernel<<<outGrid, outBlk, 0, stream>>>(
            hout, packW, b_out, out, t, xbuf);
    }
}
